// TreeAttention_2723009265703
// MI455X (gfx1250) — compile-verified, matrix operand order checked by reading, not run on hardware
//
#include <hip/hip_runtime.h>

#define E 512
#define NTOK 32768
#define HEADS 8
#define HDIM 64

typedef __bf16 bf16;
typedef __attribute__((ext_vector_type(16))) __bf16 v16bf;
typedef __attribute__((ext_vector_type(8)))  __bf16 v8bf;
typedef __attribute__((ext_vector_type(8)))  float  v8f;
typedef __attribute__((ext_vector_type(4)))  int    v4i;

// ---------------------------------------------------------------------------
// Async global->LDS staging (CDNA5 GLOBAL_LOAD_ASYNC_TO_LDS_B128, ASYNCcnt).
// Signature (from compiler diagnostic): (v4i AS1* gsrc, v4i AS3* ldst,
// imm offset, imm cpol). Falls back to a VGPR-bounce copy if absent.
// ---------------------------------------------------------------------------
#if defined(__gfx1250__) && __has_builtin(__builtin_amdgcn_global_load_async_to_lds_b128)
#define HAVE_ASYNC_LDS 1
#else
#define HAVE_ASYNC_LDS 0
#endif

__device__ __forceinline__ void async_wait_all() {
#if defined(__gfx1250__)
#if __has_builtin(__builtin_amdgcn_s_wait_asynccnt)
  __builtin_amdgcn_s_wait_asynccnt(0);
#else
  asm volatile("s_wait_asynccnt 0x0" ::: "memory");
#endif
#endif
}

// Stage 16 rows x 512 bf16 (16 KB) from global into LDS, 128b per thread/iter.
__device__ __forceinline__ void stage_tile_16xE(const bf16* __restrict__ gsrc,
                                                bf16* __restrict__ lds_dst,
                                                int tid) {
#if HAVE_ASYNC_LDS
  typedef __attribute__((address_space(1))) v4i gv4i;
  typedef __attribute__((address_space(3))) v4i lv4i;
#pragma unroll
  for (int i = 0; i < 4; ++i) {
    __builtin_amdgcn_global_load_async_to_lds_b128(
        (gv4i*)(gsrc + tid * 8 + i * 2048),
        (lv4i*)(lds_dst + tid * 8 + i * 2048),
        0, 0);
  }
  async_wait_all();
#else
  const uint4* src = (const uint4*)gsrc;
  uint4* dst = (uint4*)lds_dst;
#pragma unroll
  for (int i = 0; i < 4; ++i) dst[tid + 256 * i] = src[tid + 256 * i];
#endif
}

// ---------------------------------------------------------------------------
// Load a 16x32 bf16 WMMA operand fragment (A or symmetric-B); 16 "rows" with
// stride E. Per CDNA5 ISA 16-bit A layout:
//   lane<16 : row = lane,    K = {k0..k0+7,  k0+16..k0+23}
//   lane>=16: row = lane-16, K = {k0+8..k0+15, k0+24..k0+31}
// ---------------------------------------------------------------------------
__device__ __forceinline__ v16bf load_frag_row16(const bf16* __restrict__ base,
                                                 int k0, int lane) {
  const int row  = lane & 15;
  const int koff = (lane >> 4) << 3;   // 0 or 8
  const bf16* p = base + (size_t)row * E + k0 + koff;
  v16bf f;
  ((v8bf*)&f)[0] = *(const v8bf*)(p);        // 16B contiguous
  ((v8bf*)&f)[1] = *(const v8bf*)(p + 16);   // 16B contiguous
  return f;
}

__global__ __launch_bounds__(256) void cvt_f32_to_bf16(const float* __restrict__ s,
                                                       bf16* __restrict__ d, int n) {
  int i = (blockIdx.x * 256 + threadIdx.x) * 4;
  if (i + 3 < n) {
    const float4 v = *(const float4*)(s + i);
    d[i + 0] = (bf16)v.x;
    d[i + 1] = (bf16)v.y;
    d[i + 2] = (bf16)v.z;
    d[i + 3] = (bf16)v.w;
  }
}

// ---------------------------------------------------------------------------
// Fused: parent/child/sibling projections (WMMA) + per-token 8x8 head
// attention. One workgroup (8 wave32) owns 16 tokens; P/C/S live only in LDS.
// Each wave processes PAIRS of 16-wide N-tiles: one A-frag feeds two WMMAs
// with independent accumulators (halves ds traffic, breaks the RAW chain).
// ---------------------------------------------------------------------------
__global__ __launch_bounds__(256) void tree_proj_attn(
    const bf16* __restrict__ xb,   // (NTOK, E) bf16
    const bf16* __restrict__ Wb,   // 3 x (E, E) bf16, row-major (out, in)
    const float* __restrict__ bp, const float* __restrict__ bc,
    const float* __restrict__ bs,
    bf16* __restrict__ att)        // (NTOK, E) bf16 attention output
{
  __shared__ bf16 xs[16 * E];            // 16 KB  x tile
  __shared__ bf16 res[3][16][E];         // 48 KB  parent/child/sibling tiles

  const int blk  = blockIdx.x;           // 16-token block
  const int tid  = threadIdx.x;
  const int lane = tid & 31;
  const int wave = tid >> 5;

  stage_tile_16xE(xb + (size_t)blk * 16 * E, xs, tid);
  __syncthreads();

  // 48 tile-pairs (3 matrices x 16 pairs of two 16-wide N-tiles), 8 waves.
  for (int p = wave; p < 48; p += 8) {
    const int m  = p >> 4;               // 0=parent 1=child 2=sibling
    const int n0 = (p & 15) << 5;        // 32-wide pair base
    const bf16* W0 = Wb + (size_t)m * E * E + (size_t)n0 * E;
    const bf16* W1 = W0 + (size_t)16 * E;

    v8f acc0 = {};
    v8f acc1 = {};
#pragma unroll 4
    for (int k0 = 0; k0 < E; k0 += 32) {
      __builtin_prefetch(W0 + (size_t)(lane & 15) * E + k0 + 64, 0, 1);
      __builtin_prefetch(W1 + (size_t)(lane & 15) * E + k0 + 64, 0, 1);
      v16bf a  = load_frag_row16(xs, k0, lane);
      v16bf b0 = load_frag_row16(W0, k0, lane);
      v16bf b1 = load_frag_row16(W1, k0, lane);
      acc0 = __builtin_amdgcn_wmma_f32_16x16x32_bf16(false, a, false, b0,
                                                     (short)0, acc0, false, false);
      acc1 = __builtin_amdgcn_wmma_f32_16x16x32_bf16(false, a, false, b1,
                                                     (short)0, acc1, false, false);
    }

    const float* bias = (m == 0) ? bp : (m == 1) ? bc : bs;
    const int col0  = n0 + (lane & 15);
    const float bv0 = bias[col0];
    const float bv1 = bias[col0 + 16];
    const int rbase = (lane >> 4) << 3;  // D layout: lane>=16 holds rows 8..15
#pragma unroll
    for (int r = 0; r < 8; ++r) {
      res[m][rbase + r][col0]      = (bf16)(acc0[r] + bv0);
      res[m][rbase + r][col0 + 16] = (bf16)(acc1[r] + bv1);
    }
  }
  __syncthreads();

  // Per-token head-mixing attention: scores(8x8) over D=64, softmax, mix.
  if (tid < 128) {
    const int token = tid >> 3;
    const int h     = tid & 7;

    float sc[8];
    float mx = -3.0e38f;
#pragma unroll
    for (int t2 = 0; t2 < 8; ++t2) {
      float s = 0.f;
      for (int d = 0; d < HDIM; ++d)
        s += (float)res[0][token][h * HDIM + d] *
             (float)res[1][token][t2 * HDIM + d];
      s *= 0.125f;                       // 1/sqrt(64)
      sc[t2] = s;
      mx = fmaxf(mx, s);
    }
    float sum = 0.f;
#pragma unroll
    for (int t2 = 0; t2 < 8; ++t2) { sc[t2] = __expf(sc[t2] - mx); sum += sc[t2]; }
    const float inv = 1.f / sum;

    bf16* orow = att + ((size_t)blk * 16 + token) * E + h * HDIM;
    for (int d = 0; d < HDIM; ++d) {
      float o = 0.f;
#pragma unroll
      for (int t2 = 0; t2 < 8; ++t2)
        o += sc[t2] * (float)res[2][token][t2 * HDIM + d];
      orow[d] = (bf16)(o * inv);
    }
  }
}

// ---------------------------------------------------------------------------
// Final projection: out = att @ Wo^T + bo  (f32 output), same pairing scheme.
// ---------------------------------------------------------------------------
__global__ __launch_bounds__(256) void tree_out_proj(
    const bf16* __restrict__ att, const bf16* __restrict__ Wo,
    const float* __restrict__ bo, float* __restrict__ out)
{
  __shared__ bf16 as[16 * E];

  const int blk  = blockIdx.x;
  const int tid  = threadIdx.x;
  const int lane = tid & 31;
  const int wave = tid >> 5;

  stage_tile_16xE(att + (size_t)blk * 16 * E, as, tid);
  __syncthreads();

  for (int p = wave; p < 16; p += 8) {   // 16 pairs of two 16-wide N-tiles
    const int n0 = p << 5;
    const bf16* W0 = Wo + (size_t)n0 * E;
    const bf16* W1 = W0 + (size_t)16 * E;

    v8f acc0 = {};
    v8f acc1 = {};
#pragma unroll 4
    for (int k0 = 0; k0 < E; k0 += 32) {
      __builtin_prefetch(W0 + (size_t)(lane & 15) * E + k0 + 64, 0, 1);
      __builtin_prefetch(W1 + (size_t)(lane & 15) * E + k0 + 64, 0, 1);
      v16bf a  = load_frag_row16(as, k0, lane);
      v16bf b0 = load_frag_row16(W0, k0, lane);
      v16bf b1 = load_frag_row16(W1, k0, lane);
      acc0 = __builtin_amdgcn_wmma_f32_16x16x32_bf16(false, a, false, b0,
                                                     (short)0, acc0, false, false);
      acc1 = __builtin_amdgcn_wmma_f32_16x16x32_bf16(false, a, false, b1,
                                                     (short)0, acc1, false, false);
    }

    const int col0  = n0 + (lane & 15);
    const float bv0 = bo[col0];
    const float bv1 = bo[col0 + 16];
    const int rbase = (lane >> 4) << 3;
#pragma unroll
    for (int r = 0; r < 8; ++r) {
      const size_t rowoff = ((size_t)blk * 16 + rbase + r) * E;
      out[rowoff + col0]      = acc0[r] + bv0;
      out[rowoff + col0 + 16] = acc1[r] + bv1;
    }
  }
}

extern "C" void kernel_launch(void* const* d_in, const int* in_sizes, int n_in,
                              void* d_out, int out_size, void* d_ws, size_t ws_size,
                              hipStream_t stream) {
  (void)in_sizes; (void)n_in; (void)out_size; (void)ws_size;

  const float* x   = (const float*)d_in[0];
  const float* Wp  = (const float*)d_in[1];
  const float* bp  = (const float*)d_in[2];
  const float* Wc  = (const float*)d_in[3];
  const float* bc  = (const float*)d_in[4];
  const float* Wsb = (const float*)d_in[5];
  const float* bs  = (const float*)d_in[6];
  const float* Wo  = (const float*)d_in[7];
  const float* bo  = (const float*)d_in[8];

  // Workspace layout (bf16): xb (32 MB) | Wb 4x(E,E) (2 MB) | att (32 MB)
  bf16* xb  = (bf16*)d_ws;
  bf16* Wb  = xb + (size_t)NTOK * E;
  bf16* att = Wb + (size_t)4 * E * E;

  const int nx = NTOK * E;      // 16,777,216  (multiple of 1024)
  const int nw = E * E;         // 262,144

  cvt_f32_to_bf16<<<nx / 1024, 256, 0, stream>>>(x,   xb,                  nx);
  cvt_f32_to_bf16<<<nw / 1024, 256, 0, stream>>>(Wp,  Wb + 0 * (size_t)nw, nw);
  cvt_f32_to_bf16<<<nw / 1024, 256, 0, stream>>>(Wc,  Wb + 1 * (size_t)nw, nw);
  cvt_f32_to_bf16<<<nw / 1024, 256, 0, stream>>>(Wsb, Wb + 2 * (size_t)nw, nw);
  cvt_f32_to_bf16<<<nw / 1024, 256, 0, stream>>>(Wo,  Wb + 3 * (size_t)nw, nw);

  tree_proj_attn<<<NTOK / 16, 256, 0, stream>>>(xb, Wb, bp, bc, bs, att);
  tree_out_proj<<<NTOK / 16, 256, 0, stream>>>(att, Wb + 3 * (size_t)nw, bo,
                                               (float*)d_out);
}
